// EncoderDecoder_45896020525613
// MI455X (gfx1250) — compile-verified, matrix operand order checked by reading, not run on hardware
//
#include <hip/hip_runtime.h>
#include <cstddef>

typedef __attribute__((ext_vector_type(16))) _Float16 v16h;
typedef __attribute__((ext_vector_type(8)))  float    v8f;
// Global-address-space view of a weight-fragment array: loads through this
// lower to global_load_b128 (LOADcnt only), decoupled from LDS DScnt waits.
typedef __attribute__((address_space(1))) const v16h gv16h;

// Launder a uniform address through empty asm so weight loads are not treated
// as loop-invariant across timesteps (prevents whole-weight-set hoist +
// scratch spill), then rebuild it as an explicit global pointer.
__device__ __forceinline__ const gv16h* launder_global(const v16h* p) {
    unsigned long long u = (unsigned long long)p;
    asm volatile("" : "+s"(u));
    return (const gv16h*)u;
}

// ---------------------------------------------------------------------------
// Workspace layout (bytes). All offsets 256B aligned.
// ---------------------------------------------------------------------------
#define OFF_WGH0E   0u          // enc whh0 frags : 48*8*32*16 halves = 393216 B
#define OFF_WGI0E   393216u     // enc combined emb+wih0 frags: 48*2*512 h = 98304 B
#define OFF_WGH1E   491520u     // enc whh1 frags : 393216 B
#define OFF_WGI1E   884736u     // enc wih1 frags : 393216 B
#define OFF_WGH0D   1277952u    // dec whh0 frags : 393216 B
#define OFF_WGI0D   1671168u    // dec combined   : 98304 B
#define OFF_WGH1D   1769472u    // dec whh1 frags : 393216 B
#define OFF_WGI1D   2162688u    // dec wih1 frags : 393216 B
#define OFF_WFC     2555904u    // fc frags: 4*8*512 h = 32768 B
#define OFF_H0FIN   2588672u    // enc final h0 f16 [256][256] = 131072 B
#define OFF_H1FIN   2719744u    // enc final h1 f16 [256][256] = 131072 B
#define OFF_BCE     2850816u    // enc combined bias f32 [768] = 3072 B
#define OFF_BCD     2853888u    // dec combined bias f32 [768] = 3072 B

// ---------------------------------------------------------------------------
// Helpers
// ---------------------------------------------------------------------------
__device__ __forceinline__ float sigmoidf_(float x) {
    return 1.0f / (1.0f + __expf(-x));
}
__device__ __forceinline__ float tanhfast_(float x) {
    return 2.0f / (1.0f + __expf(-2.0f * x)) - 1.0f;
}

// Build 16x32 f16 A-fragment (per-lane v16h) from row-major f16 matrix.
// Layout (ISA 7.12.2): lanes 0-15 hold M=lane, lanes 16-31 hold M=lane-16;
// halves 0..7 -> K = khi + 0..7 ; halves 8..15 -> K = 16 + khi + 0..7,
// khi = 0 (lanes<16) or 8 (lanes>=16). Emitted as two ds_load_b128.
__device__ __forceinline__ v16h load_a_frag(const _Float16* base, int kstride,
                                            int lane, int kc) {
    int m   = lane & 15;
    int khi = (lane & 16) >> 1;   // 0 or 8
    const _Float16* row = base + m * kstride + kc * 32;
    union { v16h v; unsigned u[8]; } r;
    const unsigned* p0 = (const unsigned*)(row + khi);
    const unsigned* p1 = (const unsigned*)(row + 16 + khi);
#pragma unroll
    for (int i = 0; i < 4; i++) { r.u[i] = p0[i]; r.u[4 + i] = p1[i]; }
    return r.v;
}

// One GRU layer step for a 16-row batch slice.
// 16 waves; wave w owns gate-column-block w: N-tiles {w, w+16, w+32} = (r,z,n).
// gh = h@Whh^T + bhh (KC_H k-chunks from h_cur), gi = in@Wi^T + bi (KC_I from in).
// K-loops are software-pipelined: fragments for kc+1 are issued before the
// WMMAs of kc, so loadcnt/dscnt waits overlap with matrix math.
template <int KC_H, int KC_I>
__device__ __forceinline__ void gru_layer_step(
    const _Float16* __restrict__ h_cur,   // LDS [16][256]
    const _Float16* __restrict__ in,      // LDS [16][KC_I*32]
    _Float16* __restrict__ h_nxt,         // LDS [16][256]
    const gv16h* __restrict__ Wgh,        // frags [48][KC_H][32 lanes] (global)
    const gv16h* __restrict__ Wgi,        // frags [48][KC_I][32 lanes] (global)
    const float (&bgh)[3], const float (&bgi)[3],
    int wave, int lane) {
    v8f gh[3], gi[3];
#pragma unroll
    for (int p = 0; p < 3; p++) {
        float bh = bgh[p], bi = bgi[p];
        gh[p] = (v8f){bh, bh, bh, bh, bh, bh, bh, bh};
        gi[p] = (v8f){bi, bi, bi, bi, bi, bi, bi, bi};
    }

    // ---- GH accumulation (K = KC_H*32 from h_cur) ----
    v16h a_cur = load_a_frag(h_cur, 256, lane, 0);
    v16h b_cur[3];
#pragma unroll
    for (int p = 0; p < 3; p++)
        b_cur[p] = Wgh[((wave + 16 * p) * KC_H) * 32 + lane];
#pragma unroll
    for (int kc = 0; kc < KC_H; kc++) {
        int kcn = (kc + 1 < KC_H) ? kc + 1 : kc;   // clamped prefetch index
        v16h a_nxt = load_a_frag(h_cur, 256, lane, kcn);
        v16h b_nxt[3];
#pragma unroll
        for (int p = 0; p < 3; p++)
            b_nxt[p] = Wgh[((wave + 16 * p) * KC_H + kcn) * 32 + lane];
#pragma unroll
        for (int p = 0; p < 3; p++)
            gh[p] = __builtin_amdgcn_wmma_f32_16x16x32_f16(
                false, a_cur, false, b_cur[p], (short)0, gh[p], false, false);
        a_cur = a_nxt;
#pragma unroll
        for (int p = 0; p < 3; p++) b_cur[p] = b_nxt[p];
    }

    // ---- GI accumulation (K = KC_I*32 from in) ----
    a_cur = load_a_frag(in, KC_I * 32, lane, 0);
#pragma unroll
    for (int p = 0; p < 3; p++)
        b_cur[p] = Wgi[((wave + 16 * p) * KC_I) * 32 + lane];
#pragma unroll
    for (int kc = 0; kc < KC_I; kc++) {
        int kcn = (kc + 1 < KC_I) ? kc + 1 : kc;
        v16h a_nxt = load_a_frag(in, KC_I * 32, lane, kcn);
        v16h b_nxt[3];
#pragma unroll
        for (int p = 0; p < 3; p++)
            b_nxt[p] = Wgi[((wave + 16 * p) * KC_I + kcn) * 32 + lane];
#pragma unroll
        for (int p = 0; p < 3; p++)
            gi[p] = __builtin_amdgcn_wmma_f32_16x16x32_f16(
                false, a_cur, false, b_cur[p], (short)0, gi[p], false, false);
        a_cur = a_nxt;
#pragma unroll
        for (int p = 0; p < 3; p++) b_cur[p] = b_nxt[p];
    }

    // GRU pointwise update. C/D layout: VGPR v, lanes 0-15 -> M=v,
    // lanes 16-31 -> M=8+v; N = lane&15.
    int col   = wave * 16 + (lane & 15);
    int mbase = (lane & 16) >> 1;
#pragma unroll
    for (int vv = 0; vv < 8; vv++) {
        int   m    = mbase + vv;
        float r    = sigmoidf_(gi[0][vv] + gh[0][vv]);
        float z    = sigmoidf_(gi[1][vv] + gh[1][vv]);
        float n    = tanhfast_(gi[2][vv] + r * gh[2][vv]);
        float hold = (float)h_cur[m * 256 + col];
        h_nxt[m * 256 + col] = (_Float16)((1.0f - z) * n + z * hold);
    }
}

// ---------------------------------------------------------------------------
// Prep kernels: re-layout weights into WMMA B-fragment order.
// F[nt][kc][lane][h] = W[nt*16 + (lane&15)][kc*32 + kk(lane,h)]
// kk = ((h>>3)<<4) + ((lane>>4)<<3) + (h&7)   (mirror of A layout)
// ---------------------------------------------------------------------------
__global__ void pack_frag_kernel(const float* __restrict__ W,
                                 _Float16* __restrict__ F,
                                 int NT, int KC, int kstride) {
    int id = blockIdx.x * blockDim.x + threadIdx.x;
    if (id >= NT * KC * 512) return;
    int h    = id & 15;
    int lane = (id >> 4) & 31;
    int kc   = (id >> 9) % KC;
    int nt   = id / (KC * 512);
    int n    = nt * 16 + (lane & 15);
    int kk   = ((h >> 3) << 4) + ((lane >> 4) << 3) + (h & 7);
    F[id]    = (_Float16)W[(size_t)n * kstride + kc * 32 + kk];
}

// Combined embedding+input-gate weight: Wc = wih(768x128) @ embw(128x64).
__global__ void pack_comb_kernel(const float* __restrict__ wih,
                                 const float* __restrict__ embw,
                                 _Float16* __restrict__ F) {
    const int KC = 2;
    int id = blockIdx.x * blockDim.x + threadIdx.x;
    if (id >= 48 * KC * 512) return;
    int h    = id & 15;
    int lane = (id >> 4) & 31;
    int kc   = (id >> 9) % KC;
    int nt   = id / (KC * 512);
    int n    = nt * 16 + (lane & 15);
    int kk   = ((h >> 3) << 4) + ((lane >> 4) << 3) + (h & 7);
    int k    = kc * 32 + kk;  // 0..63
    float s = 0.0f;
    for (int j = 0; j < 128; j++) s += wih[n * 128 + j] * embw[j * 64 + k];
    F[id] = (_Float16)s;
}

// bc[g] = wih(768x128) @ embb(128) + bih[g]
__global__ void bias_comb_kernel(const float* __restrict__ wih,
                                 const float* __restrict__ embb,
                                 const float* __restrict__ bih,
                                 float* __restrict__ bc) {
    int g = blockIdx.x * blockDim.x + threadIdx.x;
    if (g >= 768) return;
    float s = bih[g];
    for (int j = 0; j < 128; j++) s += wih[g * 128 + j] * embb[j];
    bc[g] = s;
}

// ---------------------------------------------------------------------------
// Encoder: 16 WGs x 512 threads; each WG owns 16 batch rows, runs all 512
// timesteps for both GRU layers with LDS-resident double-buffered hiddens.
// Weights stream from L2 every step (set is L2-resident, ~3 MB vs 192 MB L2).
// ---------------------------------------------------------------------------
__global__ __launch_bounds__(512) void enc_gru_kernel(
    const float* __restrict__ x,  // [512][256][64]
    const v16h* __restrict__ Wgh0_, const v16h* __restrict__ Wgi0_,
    const v16h* __restrict__ Wgh1_, const v16h* __restrict__ Wgi1_,
    const float* __restrict__ bhh0, const float* __restrict__ bc0,
    const float* __restrict__ bhh1, const float* __restrict__ bih1,
    _Float16* __restrict__ h0fin, _Float16* __restrict__ h1fin) {
    __shared__ _Float16 h0[2][16 * 256];
    __shared__ _Float16 h1[2][16 * 256];
    __shared__ _Float16 xin[16 * 64];

    const int tid = threadIdx.x, wave = tid >> 5, lane = tid & 31;
    const int m0 = blockIdx.x * 16;

    for (int i = tid; i < 16 * 256; i += 512) {
        h0[0][i] = (_Float16)0.0f;
        h1[0][i] = (_Float16)0.0f;
    }
    float bgh0[3], bgi0[3], bgh1[3], bgi1[3];
#pragma unroll
    for (int p = 0; p < 3; p++) {
        int col = (wave + 16 * p) * 16 + (lane & 15);
        bgh0[p] = bhh0[col]; bgi0[p] = bc0[col];
        bgh1[p] = bhh1[col]; bgi1[p] = bih1[col];
    }
    __syncthreads();

    int cur = 0;
    for (int t = 0; t < 512; t++) {
        // Re-launder weight pointers every step (prevents cross-iteration
        // hoist/spill) while keeping them in the global address space.
        const gv16h* Wgh0 = launder_global(Wgh0_);
        const gv16h* Wgi0 = launder_global(Wgi0_);
        const gv16h* Wgh1 = launder_global(Wgh1_);
        const gv16h* Wgi1 = launder_global(Wgi1_);

        for (int i = tid; i < 16 * 64; i += 512) {
            int m = i >> 6, c = i & 63;
            xin[i] = (_Float16)x[((size_t)t * 256 + m0 + m) * 64 + c];
        }
        __syncthreads();
        gru_layer_step<8, 2>(h0[cur], xin, h0[cur ^ 1], Wgh0, Wgi0,
                             bgh0, bgi0, wave, lane);
        __syncthreads();
        gru_layer_step<8, 8>(h1[cur], h0[cur ^ 1], h1[cur ^ 1], Wgh1, Wgi1,
                             bgh1, bgi1, wave, lane);
        __syncthreads();
        cur ^= 1;
    }
    for (int i = tid; i < 16 * 256; i += 512) {
        int m = i >> 8, c = i & 255;
        h0fin[(size_t)(m0 + m) * 256 + c] = h0[cur][i];
        h1fin[(size_t)(m0 + m) * 256 + c] = h1[cur][i];
    }
}

// ---------------------------------------------------------------------------
// Decoder: 24 autoregressive steps; fc output feeds back as next input.
// ---------------------------------------------------------------------------
__global__ __launch_bounds__(512) void dec_gru_kernel(
    const float* __restrict__ x,
    const _Float16* __restrict__ h0fin, const _Float16* __restrict__ h1fin,
    const v16h* __restrict__ Wgh0_, const v16h* __restrict__ Wgi0_,
    const v16h* __restrict__ Wgh1_, const v16h* __restrict__ Wgi1_,
    const v16h* __restrict__ Wfc_,
    const float* __restrict__ bhh0, const float* __restrict__ bc0,
    const float* __restrict__ bhh1, const float* __restrict__ bih1,
    const float* __restrict__ fcb, float* __restrict__ out) {
    __shared__ _Float16 h0[2][16 * 256];
    __shared__ _Float16 h1[2][16 * 256];
    __shared__ _Float16 din[16 * 64];

    const int tid = threadIdx.x, wave = tid >> 5, lane = tid & 31;
    const int m0 = blockIdx.x * 16;

    for (int i = tid; i < 16 * 256; i += 512) {
        int m = i >> 8, c = i & 255;
        h0[0][i] = h0fin[(size_t)(m0 + m) * 256 + c];
        h1[0][i] = h1fin[(size_t)(m0 + m) * 256 + c];
    }
    for (int i = tid; i < 16 * 64; i += 512) {
        int m = i >> 6, c = i & 63;
        din[i] = (_Float16)x[((size_t)511 * 256 + m0 + m) * 64 + c];
    }
    float bgh0[3], bgi0[3], bgh1[3], bgi1[3];
#pragma unroll
    for (int p = 0; p < 3; p++) {
        int col = (wave + 16 * p) * 16 + (lane & 15);
        bgh0[p] = bhh0[col]; bgi0[p] = bc0[col];
        bgh1[p] = bhh1[col]; bgi1[p] = bih1[col];
    }
    float fbias = (wave < 4) ? fcb[wave * 16 + (lane & 15)] : 0.0f;
    __syncthreads();

    int cur = 0;
    for (int t = 0; t < 24; t++) {
        const gv16h* Wgh0 = launder_global(Wgh0_);
        const gv16h* Wgi0 = launder_global(Wgi0_);
        const gv16h* Wgh1 = launder_global(Wgh1_);
        const gv16h* Wgi1 = launder_global(Wgi1_);
        const gv16h* Wfc  = launder_global(Wfc_);

        gru_layer_step<8, 2>(h0[cur], din, h0[cur ^ 1], Wgh0, Wgi0,
                             bgh0, bgi0, wave, lane);
        __syncthreads();
        gru_layer_step<8, 8>(h1[cur], h0[cur ^ 1], h1[cur ^ 1], Wgh1, Wgi1,
                             bgh1, bgi1, wave, lane);
        __syncthreads();
        // fc: o = h1_new @ fc_w^T + fc_b ; N=64 -> 4 tiles on waves 0..3.
        if (wave < 4) {
            v8f acc = (v8f){fbias, fbias, fbias, fbias,
                            fbias, fbias, fbias, fbias};
            v16h a_cur = load_a_frag(h1[cur ^ 1], 256, lane, 0);
            v16h b_cur = Wfc[(wave * 8) * 32 + lane];
#pragma unroll
            for (int kc = 0; kc < 8; kc++) {
                int kcn = (kc + 1 < 8) ? kc + 1 : kc;
                v16h a_nxt = load_a_frag(h1[cur ^ 1], 256, lane, kcn);
                v16h b_nxt = Wfc[(wave * 8 + kcn) * 32 + lane];
                acc = __builtin_amdgcn_wmma_f32_16x16x32_f16(
                    false, a_cur, false, b_cur, (short)0, acc, false, false);
                a_cur = a_nxt;
                b_cur = b_nxt;
            }
            int col   = wave * 16 + (lane & 15);
            int mbase = (lane & 16) >> 1;
#pragma unroll
            for (int vv = 0; vv < 8; vv++) {
                int   m = mbase + vv;
                float o = acc[vv];
                out[((size_t)t * 256 + m0 + m) * 64 + col] = o;
                din[m * 64 + col] = (_Float16)o;
            }
        }
        __syncthreads();
        cur ^= 1;
    }
}

// ---------------------------------------------------------------------------
// Host launcher
// ---------------------------------------------------------------------------
extern "C" void kernel_launch(void* const* d_in, const int* in_sizes, int n_in,
                              void* d_out, int out_size, void* d_ws,
                              size_t ws_size, hipStream_t stream) {
    const float* x        = (const float*)d_in[0];
    const float* eembw    = (const float*)d_in[1];
    const float* eembb    = (const float*)d_in[2];
    const float* ewih0    = (const float*)d_in[3];
    const float* ewhh0    = (const float*)d_in[4];
    const float* ebih0    = (const float*)d_in[5];
    const float* ebhh0    = (const float*)d_in[6];
    const float* ewih1    = (const float*)d_in[7];
    const float* ewhh1    = (const float*)d_in[8];
    const float* ebih1    = (const float*)d_in[9];
    const float* ebhh1    = (const float*)d_in[10];
    const float* dembw    = (const float*)d_in[11];
    const float* dembb    = (const float*)d_in[12];
    const float* dwih0    = (const float*)d_in[13];
    const float* dwhh0    = (const float*)d_in[14];
    const float* dbih0    = (const float*)d_in[15];
    const float* dbhh0    = (const float*)d_in[16];
    const float* dwih1    = (const float*)d_in[17];
    const float* dwhh1    = (const float*)d_in[18];
    const float* dbih1    = (const float*)d_in[19];
    const float* dbhh1    = (const float*)d_in[20];
    const float* fcw      = (const float*)d_in[21];
    const float* fcb      = (const float*)d_in[22];
    float* out = (float*)d_out;

    char* ws = (char*)d_ws;
    _Float16* Wgh0e = (_Float16*)(ws + OFF_WGH0E);
    _Float16* Wgi0e = (_Float16*)(ws + OFF_WGI0E);
    _Float16* Wgh1e = (_Float16*)(ws + OFF_WGH1E);
    _Float16* Wgi1e = (_Float16*)(ws + OFF_WGI1E);
    _Float16* Wgh0d = (_Float16*)(ws + OFF_WGH0D);
    _Float16* Wgi0d = (_Float16*)(ws + OFF_WGI0D);
    _Float16* Wgh1d = (_Float16*)(ws + OFF_WGH1D);
    _Float16* Wgi1d = (_Float16*)(ws + OFF_WGI1D);
    _Float16* Wfc   = (_Float16*)(ws + OFF_WFC);
    _Float16* h0fin = (_Float16*)(ws + OFF_H0FIN);
    _Float16* h1fin = (_Float16*)(ws + OFF_H1FIN);
    float*    bcE   = (float*)(ws + OFF_BCE);
    float*    bcD   = (float*)(ws + OFF_BCD);

    const int THR = 256;
    auto nb = [](int n, int t) { return (n + t - 1) / t; };

    // --- weight packing ---
    pack_frag_kernel<<<nb(48 * 8 * 512, THR), THR, 0, stream>>>(ewhh0, Wgh0e, 48, 8, 256);
    pack_comb_kernel<<<nb(48 * 2 * 512, THR), THR, 0, stream>>>(ewih0, eembw, Wgi0e);
    pack_frag_kernel<<<nb(48 * 8 * 512, THR), THR, 0, stream>>>(ewhh1, Wgh1e, 48, 8, 256);
    pack_frag_kernel<<<nb(48 * 8 * 512, THR), THR, 0, stream>>>(ewih1, Wgi1e, 48, 8, 256);
    pack_frag_kernel<<<nb(48 * 8 * 512, THR), THR, 0, stream>>>(dwhh0, Wgh0d, 48, 8, 256);
    pack_comb_kernel<<<nb(48 * 2 * 512, THR), THR, 0, stream>>>(dwih0, dembw, Wgi0d);
    pack_frag_kernel<<<nb(48 * 8 * 512, THR), THR, 0, stream>>>(dwhh1, Wgh1d, 48, 8, 256);
    pack_frag_kernel<<<nb(48 * 8 * 512, THR), THR, 0, stream>>>(dwih1, Wgi1d, 48, 8, 256);
    pack_frag_kernel<<<nb(4 * 8 * 512, THR), THR, 0, stream>>>(fcw, Wfc, 4, 8, 256);
    bias_comb_kernel<<<3, 256, 0, stream>>>(ewih0, eembb, ebih0, bcE);
    bias_comb_kernel<<<3, 256, 0, stream>>>(dwih0, dembb, dbih0, bcD);

    // --- encoder recurrence (batch-parallel over 16 WGs) ---
    enc_gru_kernel<<<16, 512, 0, stream>>>(
        x, (const v16h*)Wgh0e, (const v16h*)Wgi0e,
        (const v16h*)Wgh1e, (const v16h*)Wgi1e,
        ebhh0, bcE, ebhh1, ebih1, h0fin, h1fin);

    // --- autoregressive decoder ---
    dec_gru_kernel<<<16, 512, 0, stream>>>(
        x, h0fin, h1fin,
        (const v16h*)Wgh0d, (const v16h*)Wgi0d,
        (const v16h*)Wgh1d, (const v16h*)Wgi1d, (const v16h*)Wfc,
        dbhh0, bcD, dbhh1, dbih1, fcb, out);

    (void)in_sizes; (void)n_in; (void)out_size; (void)ws_size;
}
